// BFFN_83949430767982
// MI455X (gfx1250) — compile-verified, matrix operand order checked by reading, not run on hardware
//
#include <hip/hip_runtime.h>
#include <hip/hip_bf16.h>

// CDNA5 / gfx1250: wave32, v_wmma_f32_16x16x32_bf16 (f32 accumulate),
// global_load_async_to_lds_b128 for the tile pipeline (ASYNCcnt-tracked DMA-style copy).
typedef __attribute__((ext_vector_type(16))) __bf16 v16bf;
typedef __attribute__((ext_vector_type(8)))  float  v8f;

#define TM   128      // block tile M
#define TN   256      // block tile N
#define TK   32       // K step (WMMA K)
#define LDSS 40       // padded LDS row stride (32 + 8), rows stay 16B aligned (80B)

// Async global -> LDS 16-byte copy, GVS mode: mem = saddr(SGPR64) + voffset(VGPR i32).
// No VGPR data staging; tracked with ASYNCcnt.
__device__ __forceinline__ void async_copy_b128(unsigned lds_off, const void* sbase,
                                                unsigned byte_off) {
    asm volatile("global_load_async_to_lds_b128 %0, %1, %2"
                 :
                 : "v"(lds_off), "v"(byte_off), "s"(sbase)
                 : "memory");
}

__device__ __forceinline__ void wait_async0() {
    asm volatile("s_wait_asynccnt 0x0" ::: "memory");
}

// -------- elementwise helpers --------

__global__ __launch_bounds__(256)
void f32_to_bf16_vec4(const float* __restrict__ in, __bf16* __restrict__ out, int n4) {
    int i = blockIdx.x * 256 + threadIdx.x;
    if (i < n4) {
        float4 f = ((const float4*)in)[i];
        union { __bf16 h[4]; unsigned long long u; } o;
        o.h[0] = (__bf16)f.x; o.h[1] = (__bf16)f.y;
        o.h[2] = (__bf16)f.z; o.h[3] = (__bf16)f.w;
        ((unsigned long long*)out)[i] = o.u;
    }
}

// in: f32 [R][C] row-major  ->  out: bf16 [C][R] (transposed)
__global__ __launch_bounds__(256)
void f32_to_bf16_transpose(const float* __restrict__ in, __bf16* __restrict__ out,
                           int R, int C) {
    int idx = blockIdx.x * 256 + threadIdx.x;
    if (idx < R * C) {
        int r = idx / C, c = idx % C;
        out[(long long)c * R + r] = (__bf16)in[idx];
    }
}

__global__ __launch_bounds__(256)
void gate_mul_bf16(const __bf16* __restrict__ a, const __bf16* __restrict__ b,
                   __bf16* __restrict__ c, int n2) {
    int i = blockIdx.x * 256 + threadIdx.x;
    if (i < n2) {
        union U { __bf16 h[2]; unsigned int u; };
        U x, y, z;
        x.u = ((const unsigned int*)a)[i];
        y.u = ((const unsigned int*)b)[i];
        z.h[0] = (__bf16)((float)x.h[0] * (float)y.h[0]);
        z.h[1] = (__bf16)((float)x.h[1] * (float)y.h[1]);
        ((unsigned int*)c)[i] = z.u;
    }
}

// -------- WMMA bf16 GEMM --------
// C[M,N] = A @ B with A row-major [M,K] (lda), B supplied TRANSPOSED: Bt [N,K] (ldb).
// Both LDS fills are straight b128 row copies done by async global->LDS.
// CTRANS=0: C row-major (ldc), scalar stores. CTRANS=1: Ct[N,M] (ldc), packed b128 stores.
// Block: 256 threads = 8 waves, wave grid 2(M) x 4(N), 64x64 per wave => 4x4 WMMA accs.
// Double-buffered LDS; prefetch via ASYNCcnt-tracked copies overlapped with WMMA.
template<bool CTRANS, bool OUTF32>
__global__ __launch_bounds__(256)
void gemm_wmma_bf16(const __bf16* __restrict__ Ag, const __bf16* __restrict__ Btg,
                    void* __restrict__ Cg, int M, int N, int K,
                    int lda, int ldb, int ldc,
                    long long sA, long long sB, long long sC)
{
    __shared__ __bf16 As[2][TM * LDSS];   // [m][k]
    __shared__ __bf16 Bs[2][TN * LDSS];   // [n][k]

    const int tid  = threadIdx.x;
    const int lane = tid & 31;
    const int wave = tid >> 5;

    const __bf16* A  = Ag  + (long long)blockIdx.z * sA;
    const __bf16* Bt = Btg + (long long)blockIdx.z * sB;

    const int m0 = blockIdx.y * TM;
    const int n0 = blockIdx.x * TN;
    const int wm = (wave >> 2) * 64;   // wave M offset (2 waves along M)
    const int wn = (wave & 3) * 64;    // wave N offset (4 waves along N)
    const int r  = lane & 15;
    const int h  = lane >> 4;

    // raw 32-bit LDS byte offsets (low dword of the LDS-aperture generic address)
    const unsigned asA = (unsigned)(size_t)(void*)&As[0][0];
    const unsigned asB = (unsigned)(size_t)(void*)&Bs[0][0];

    // this thread's 2 A-chunks and 4 B-chunks (16B each) of the 128x32 / 256x32 tiles
    int arow[2], aq[2], brow[4], bq[4];
    #pragma unroll
    for (int it = 0; it < 2; ++it) {
        int idx = it * 256 + tid; arow[it] = idx >> 2; aq[it] = idx & 3;
    }
    #pragma unroll
    for (int it = 0; it < 4; ++it) {
        int idx = it * 256 + tid; brow[it] = idx >> 2; bq[it] = idx & 3;
    }

    v8f acc[4][4] = {};

    // ---- prologue: async-fill buffer 0 with tile k0=0 ----
    #pragma unroll
    for (int it = 0; it < 2; ++it)
        async_copy_b128(asA + (unsigned)((arow[it] * LDSS + aq[it] * 8) * 2), A,
                        (unsigned)(((long long)(m0 + arow[it]) * lda + aq[it] * 8) * 2));
    #pragma unroll
    for (int it = 0; it < 4; ++it)
        async_copy_b128(asB + (unsigned)((brow[it] * LDSS + bq[it] * 8) * 2), Bt,
                        (unsigned)(((long long)(n0 + brow[it]) * ldb + bq[it] * 8) * 2));
    wait_async0();
    __syncthreads();

    int buf = 0;
    for (int k0 = 0; k0 < K; k0 += TK) {
        const bool pref = (k0 + TK) < K;

        // ---- launch async prefetch of the next tile into buf^1 (runs under WMMA) ----
        if (pref) {
            const int kk = k0 + TK;
            const unsigned ab = (unsigned)((buf ^ 1) * (TM * LDSS) * 2);
            const unsigned bb = (unsigned)((buf ^ 1) * (TN * LDSS) * 2);
            #pragma unroll
            for (int it = 0; it < 2; ++it)
                async_copy_b128(asA + ab + (unsigned)((arow[it] * LDSS + aq[it] * 8) * 2), A,
                                (unsigned)(((long long)(m0 + arow[it]) * lda + kk + aq[it] * 8) * 2));
            #pragma unroll
            for (int it = 0; it < 4; ++it)
                async_copy_b128(asB + bb + (unsigned)((brow[it] * LDSS + bq[it] * 8) * 2), Bt,
                                (unsigned)(((long long)(n0 + brow[it]) * ldb + kk + bq[it] * 8) * 2));
        }

        // ---- fragments per CDNA5 16-bit A/B layout: k pair base for VGPR v,
        //      half h: (v<4) ? h*8+2v : 16+h*8+2(v-4) ----
        union FB { v16bf v; unsigned int u[8]; };
        FB af[4], bfr[4];
        #pragma unroll
        for (int j = 0; j < 4; ++j) {
            const int n = wn + j * 16 + r;
            #pragma unroll
            for (int v = 0; v < 8; ++v) {
                const int kb = (v < 4) ? (h * 8 + 2 * v) : (16 + h * 8 + 2 * (v - 4));
                bfr[j].u[v] = *(const unsigned int*)&Bs[buf][n * LDSS + kb];
            }
        }
        #pragma unroll
        for (int i = 0; i < 4; ++i) {
            const int m = wm + i * 16 + r;
            #pragma unroll
            for (int v = 0; v < 8; ++v) {
                const int kb = (v < 4) ? (h * 8 + 2 * v) : (16 + h * 8 + 2 * (v - 4));
                af[i].u[v] = *(const unsigned int*)&As[buf][m * LDSS + kb];
            }
        }

        // ---- 16 WMMAs per wave per K-step ----
        #pragma unroll
        for (int i = 0; i < 4; ++i)
            #pragma unroll
            for (int j = 0; j < 4; ++j)
                acc[i][j] = __builtin_amdgcn_wmma_f32_16x16x32_bf16(
                    false, af[i].v, false, bfr[j].v, (short)0, acc[i][j], false, false);

        // ---- drain asyncs, swap buffers ----
        wait_async0();
        __syncthreads();
        buf ^= 1;
    }

    // ---- epilogue. D layout: lane col = r, VGPR v -> row = v + 8*h (8 consecutive) ----
    if (CTRANS) {
        // Ct[n][m]: each lane stores 8 consecutive m as one packed b128
        __bf16* Ct = (__bf16*)Cg + (long long)blockIdx.z * sC;
        #pragma unroll
        for (int i = 0; i < 4; ++i)
            #pragma unroll
            for (int j = 0; j < 4; ++j) {
                const int nn = n0 + wn + j * 16 + r;
                const int mb = m0 + wm + i * 16 + h * 8;
                union { __bf16 hh[8]; uint4 u; } o;
                #pragma unroll
                for (int v = 0; v < 8; ++v) o.hh[v] = (__bf16)acc[i][j][v];
                *(uint4*)&Ct[(long long)nn * ldc + mb] = o.u;
            }
    } else {
        #pragma unroll
        for (int i = 0; i < 4; ++i)
            #pragma unroll
            for (int j = 0; j < 4; ++j) {
                const int nn = n0 + wn + j * 16 + r;
                #pragma unroll
                for (int v = 0; v < 8; ++v) {
                    const int mm = m0 + wm + i * 16 + h * 8 + v;
                    const long long off = (long long)mm * ldc + nn;
                    if (OUTF32)
                        (((float*)Cg) + (long long)blockIdx.z * sC)[off] = acc[i][j][v];
                    else
                        (((__bf16*)Cg) + (long long)blockIdx.z * sC)[off] = (__bf16)acc[i][j][v];
                }
            }
    }
}

// -------- host launcher --------

extern "C" void kernel_launch(void* const* d_in, const int* in_sizes, int n_in,
                              void* d_out, int out_size, void* d_ws, size_t ws_size,
                              hipStream_t stream) {
    (void)in_sizes; (void)n_in; (void)out_size; (void)ws_size;

    const float* x_real = (const float*)d_in[0];
    const float* x_imag = (const float*)d_in[1];
    const float* w_qr   = (const float*)d_in[2];
    const float* w_qi   = (const float*)d_in[3];
    const float* w_k    = (const float*)d_in[4];
    const float* w_v    = (const float*)d_in[5];
    float* out = (float*)d_out;

    constexpr long long BATCH = 4, NSEQ = 4096, D = 1024;
    constexpr long long BND = BATCH * NSEQ * D;   // 16,777,216
    constexpr long long DD  = D * D;              // 1,048,576
    const int MB = (int)(BATCH * NSEQ);           // 16384

    // workspace layout (bf16)
    __bf16* Xr   = (__bf16*)d_ws;      // [B*N][D]
    __bf16* Xi   = Xr   + BND;
    __bf16* WqrT = Xi   + BND;         // [e][d]
    __bf16* WqiT = WqrT + DD;
    __bf16* WkT  = WqiT + DD;
    __bf16* WvT  = WkT  + DD;
    __bf16* QR   = WvT  + DD;          // [B*N][e]
    __bf16* QI   = QR   + BND;
    __bf16* Kt   = QI   + BND;         // [d][B*N]   (K projection, transposed)
    __bf16* Vt   = Kt   + BND;         // [e][B*N]   (V projection, transposed)
    __bf16* KVt  = Vt   + BND;         // [B][e][d]  (kv^T per batch)

    dim3 blk(256);

    // 1) casts (+ weight transpose) — bandwidth-trivial pre-pass
    f32_to_bf16_vec4<<<(int)(BND / 4 / 256), blk, 0, stream>>>(x_real, Xr, (int)(BND / 4));
    f32_to_bf16_vec4<<<(int)(BND / 4 / 256), blk, 0, stream>>>(x_imag, Xi, (int)(BND / 4));
    f32_to_bf16_transpose<<<(int)(DD / 256), blk, 0, stream>>>(w_qr, WqrT, (int)D, (int)D);
    f32_to_bf16_transpose<<<(int)(DD / 256), blk, 0, stream>>>(w_qi, WqiT, (int)D, (int)D);
    f32_to_bf16_transpose<<<(int)(DD / 256), blk, 0, stream>>>(w_k,  WkT,  (int)D, (int)D);
    f32_to_bf16_transpose<<<(int)(DD / 256), blk, 0, stream>>>(w_v,  WvT,  (int)D, (int)D);

    // 2) projections: [16384,1024] x [1024,1024]
    dim3 gproj(D / TN, MB / TM, 1);   // (4, 128, 1)
    gemm_wmma_bf16<false, false><<<gproj, blk, 0, stream>>>(
        Xr, WqrT, QR, MB, (int)D, (int)D, (int)D, (int)D, (int)D, 0, 0, 0);
    gemm_wmma_bf16<false, false><<<gproj, blk, 0, stream>>>(
        Xi, WqiT, QI, MB, (int)D, (int)D, (int)D, (int)D, (int)D, 0, 0, 0);
    // K and V written transposed: Kt[d][B*N], Vt[e][B*N] (ldc = MB)
    gemm_wmma_bf16<true, false><<<gproj, blk, 0, stream>>>(
        Xr, WkT, Kt, MB, (int)D, (int)D, (int)D, (int)D, MB, 0, 0, 0);
    gemm_wmma_bf16<true, false><<<gproj, blk, 0, stream>>>(
        Xi, WvT, Vt, MB, (int)D, (int)D, (int)D, (int)D, MB, 0, 0, 0);

    // 3) gate: Q = QR * QI (in place)
    gate_mul_bf16<<<(int)(BND / 2 / 256), blk, 0, stream>>>(QR, QI, QR, (int)(BND / 2));

    // 4) kv[b] = K[b]^T @ V[b].  A = Kt (lda=MB, batch col-offset NSEQ),
    //    Bt = Vt (ldb=MB, batch col-offset NSEQ). Output transposed KVt[b][e][d], ldc=D.
    dim3 gkv(D / TN, D / TM, (unsigned)BATCH);   // (4, 8, 4)
    gemm_wmma_bf16<true, false><<<gkv, blk, 0, stream>>>(
        Kt, Vt, KVt, (int)D, (int)D, (int)NSEQ, MB, MB, (int)D, NSEQ, NSEQ, DD);

    // 5) out[b] = Q[b] @ kv[b].  A = QR (gated), Bt = KVt[b] ([e][d], ldb=D), f32 out.
    dim3 gout(D / TN, NSEQ / TM, (unsigned)BATCH);  // (4, 32, 4)
    gemm_wmma_bf16<false, true><<<gout, blk, 0, stream>>>(
        QR, KVt, out, (int)NSEQ, (int)D, (int)D, (int)D, (int)D, (int)D,
        NSEQ * D, DD, NSEQ * D);
}
